// DGCNN_cls_semseg_86268713108067
// MI455X (gfx1250) — matrix operand order repaired and checked by reading, not compiled, not run on hardware
//
#include <hip/hip_runtime.h>
#include <cstdint>
#include <cstddef>

// ---------------- constants ----------------
#define NPTS 2048
#define KNN  20
#define NB   4
#define NKM  (NPTS * KNN)   // 40960 columns for edge convs
#define TDM_CHUNK 4096      // floats per TDM copy chunk (16 KB)

typedef __attribute__((ext_vector_type(16))) __bf16 v16bf;
typedef __attribute__((ext_vector_type(8)))  float  v8f;
typedef __attribute__((ext_vector_type(2)))  float  v2f;
typedef __attribute__((ext_vector_type(4)))  unsigned int v4u;
typedef __attribute__((ext_vector_type(8)))  int    v8i;
typedef __attribute__((ext_vector_type(4)))  int    v4i;

__device__ __forceinline__ float lrelu_f(float v) { return fmaxf(v, 0.2f * v); }

// ---------------- BN param folding ----------------
__global__ void bn_prep_k(const float* g, const float* b, const float* m,
                          const float* v, float* scale, float* shift, int C) {
    int i = blockIdx.x * blockDim.x + threadIdx.x;
    if (i < C) {
        float s = g[i] * rsqrtf(v[i] + 1e-5f);
        scale[i] = s;
        shift[i] = b[i] - m[i] * s;
    }
}

__global__ void fold_bias_k(const float* scale, const float* shift,
                            const float* bias, float* out, int C) {
    int i = blockIdx.x * blockDim.x + threadIdx.x;
    if (i < C) out[i] = shift[i] + bias[i] * scale[i];
}

// ---------------- Gram matrix via V_WMMA_F32_16X16X4_F32 ----------------
// X: (NB, C, NPTS) f32  ->  G[b,n,m] = sum_c X[b,c,n]*X[b,c,m]
__global__ void gram_wmma_k(const float* __restrict__ X, float* __restrict__ G, int C) {
    const int lane = threadIdx.x & 31;
    const int wid  = threadIdx.x >> 5;
    const int tiles = NPTS / 16;                       // 128
    long t = (long)blockIdx.x * (blockDim.x >> 5) + wid;
    long total = (long)NB * tiles * tiles;
    if (t >= total) return;
    int b   = (int)(t / ((long)tiles * tiles));
    long r  = t - (long)b * tiles * tiles;
    int tn  = (int)(r / tiles);
    int tm  = (int)(r % tiles);

    const float* Xb = X + (long)b * C * NPTS;
    const int row   = tn * 16 + (lane & 15);
    const int col   = tm * 16 + (lane & 15);
    const int khalf = (lane >> 4) * 2;                 // f32 A/B fragment: K = khalf + i

    const float* pr = Xb + row;   // stride NPTS per channel
    const float* pc = Xb + col;

    v8f acc = {};
    int c0 = 0;
    const int Cfull = C & ~3;
    // unguarded main loop (C = 64, 128), unrolled x2
    for (; c0 + 8 <= Cfull; c0 += 8) {
        v2f a0, b0, a1, b1;
        long k0 = (long)(c0 + khalf) * NPTS;
        a0[0] = pr[k0];          a0[1] = pr[k0 + NPTS];
        b0[0] = pc[k0];          b0[1] = pc[k0 + NPTS];
        long k1 = k0 + 4L * NPTS;
        a1[0] = pr[k1];          a1[1] = pr[k1 + NPTS];
        b1[0] = pc[k1];          b1[1] = pc[k1 + NPTS];
        acc = __builtin_amdgcn_wmma_f32_16x16x4_f32(false, a0, false, b0, (short)0, acc, false, false);
        acc = __builtin_amdgcn_wmma_f32_16x16x4_f32(false, a1, false, b1, (short)0, acc, false, false);
    }
    for (; c0 + 4 <= Cfull; c0 += 4) {
        v2f a, bb;
        long k0 = (long)(c0 + khalf) * NPTS;
        a[0]  = pr[k0];  a[1]  = pr[k0 + NPTS];
        bb[0] = pc[k0];  bb[1] = pc[k0 + NPTS];
        acc = __builtin_amdgcn_wmma_f32_16x16x4_f32(false, a, false, bb, (short)0, acc, false, false);
    }
    if (c0 < C) {                                      // tail (only C=3): clamp + select
        v2f a, bb;
        int ka = c0 + khalf, kb = c0 + khalf + 1;
        int kac = ka < C ? ka : C - 1;
        int kbc = kb < C ? kb : C - 1;
        float a0 = pr[(long)kac * NPTS], a1 = pr[(long)kbc * NPTS];
        float c0v = pc[(long)kac * NPTS], c1v = pc[(long)kbc * NPTS];
        a[0]  = (ka < C) ? a0 : 0.f;   a[1]  = (kb < C) ? a1 : 0.f;
        bb[0] = (ka < C) ? c0v : 0.f;  bb[1] = (kb < C) ? c1v : 0.f;
        acc = __builtin_amdgcn_wmma_f32_16x16x4_f32(false, a, false, bb, (short)0, acc, false, false);
    }

    float* Gb = G + (long)b * NPTS * NPTS;
    const int hi = lane >> 4;
#pragma unroll
    for (int rr = 0; rr < 8; ++rr) {
        int orow = tn * 16 + rr + 8 * hi;
        Gb[(long)orow * NPTS + col] = acc[rr];
    }
}

__global__ void gram_diag_k(const float* __restrict__ G, float* __restrict__ xx) {
    long i = (long)blockIdx.x * blockDim.x + threadIdx.x;
    if (i >= (long)NB * NPTS) return;
    long b = i >> 11, n = i & (NPTS - 1);
    xx[i] = G[(b * NPTS + n) * NPTS + n];
}

// ---------------- top-k(20) neighbor selection: one wave per row ----------------
__global__ void knn_topk_k(const float* __restrict__ G, const float* __restrict__ xx,
                           int* __restrict__ idx) {
    const int bn = blockIdx.x;
    const int b  = bn / NPTS, n = bn % NPTS;
    const float* grow = G + ((long)b * NPTS + n) * NPTS;
    const float* xxb  = xx + (long)b * NPTS;
    const float  xxn  = xxb[n];
    const int lane = threadIdx.x;

    float vals[KNN]; int ids[KNN];
#pragma unroll
    for (int i = 0; i < KNN; ++i) { vals[i] = -3.4e38f; ids[i] = 0; }

    for (int m = lane; m < NPTS; m += 32) {
        float pd = 2.f * grow[m] - xxn - xxb[m];       // = -||x_n - x_m||^2
        if (pd > vals[KNN - 1]) {
            vals[KNN - 1] = pd; ids[KNN - 1] = m;
#pragma unroll
            for (int j = KNN - 1; j > 0; --j) {
                if (vals[j] > vals[j - 1]) {
                    float tv = vals[j]; vals[j] = vals[j - 1]; vals[j - 1] = tv;
                    int   ti = ids[j];  ids[j]  = ids[j - 1];  ids[j - 1]  = ti;
                }
            }
        }
    }
    __shared__ float sv[32 * KNN];
    __shared__ int   si[32 * KNN];
#pragma unroll
    for (int i = 0; i < KNN; ++i) { sv[lane * KNN + i] = vals[i]; si[lane * KNN + i] = ids[i]; }
    __syncthreads();
    if (lane == 0) {
        float bv[KNN]; int bi[KNN];
#pragma unroll
        for (int i = 0; i < KNN; ++i) { bv[i] = -3.4e38f; bi[i] = 0; }
        for (int e = 0; e < 32 * KNN; ++e) {
            float v = sv[e];
            if (v > bv[KNN - 1]) {
                bv[KNN - 1] = v; bi[KNN - 1] = si[e];
#pragma unroll
                for (int j = KNN - 1; j > 0; --j) {
                    if (bv[j] > bv[j - 1]) {
                        float tv = bv[j]; bv[j] = bv[j - 1]; bv[j - 1] = tv;
                        int   ti = bi[j]; bi[j] = bi[j - 1]; bi[j - 1] = ti;
                    }
                }
            }
        }
        int* op = idx + ((long)b * NPTS + n) * KNN;
#pragma unroll
        for (int i = 0; i < KNN; ++i) op[i] = bi[i];
    }
}

// ---------------- generic GEMM via V_WMMA_F32_16X16X32_BF16 ----------------
// Out[b,o,m] = epi( sum_k W[o,k] * Bmat[b,k,m] )
// BMODE 0 (PLAIN): Bmat = Xsrc (NB, Kd, M)
// BMODE 1 (EDGE):  M = NPTS*KNN, Kd = 2*Cin, m -> (n, kk):
//                  k < Cin : Xsrc[k, idx[n,kk]] - Xsrc[k, n]
//                  k >= Cin: Xsrc[k-Cin, n]            (Xsrc: (NB, Cin, NPTS))
// EPI 0: none   EPI 1: y = lrelu(y*scale[o]+shift[o])   EPI 2: y += scale[o] (bias)
// BOUND false: requires O%16==0, Kd%32==0, tilesM%MT==0 -> fully unguarded hot path
// MT: 16-wide M tiles per wave (register blocking; A fragment reused MT times)
template <int BMODE, int EPI, bool BOUND, int MT>
__global__ void gemm_bf16_k(const float* __restrict__ W, const float* __restrict__ Xsrc,
                            const int* __restrict__ idx, float* __restrict__ Out,
                            int O, int Kd, int M, int Cin,
                            long xBatchStride, long outBatchStride,
                            const float* __restrict__ scale, const float* __restrict__ shift) {
    const int lane = threadIdx.x & 31;
    const int wid  = threadIdx.x >> 5;
    const int tilesO = (O + 15) >> 4;
    const int tgM    = ((M + 15) >> 4) / MT;           // host guarantees divisibility
    long t = (long)blockIdx.x * (blockDim.x >> 5) + wid;
    long total = (long)NB * tilesO * tgM;
    if (t >= total) return;
    int b  = (int)(t / ((long)tilesO * tgM));
    long r = t - (long)b * tilesO * tgM;
    int to = (int)(r / tgM), tmg = (int)(r % tgM);

    const float* Xb  = Xsrc + (long)b * xBatchStride;
    const int rowA   = (to << 4) + (lane & 15);
    const int colB   = (tmg * MT << 4) + (lane & 15);  // base column; sub-tiles at +16*mt
    const int hi     = lane >> 4;
    const int kbA    = 16 * hi;   // bf16 A fragment K-base
    const int kbB    = 16 * hi;   // bf16 B fragment K-base

    // edge-mode loop invariants (hoisted): neighbor index per sub-tile column
    int en[MT], nbi[MT];
    if (BMODE == 1) {
#pragma unroll
        for (int mt = 0; mt < MT; ++mt) {
            int col = colB + 16 * mt;
            int cc = BOUND ? (col < M ? col : M - 1) : col;
            en[mt]  = cc / KNN;
            int ekk = cc - en[mt] * KNN;
            nbi[mt] = idx[(long)b * NPTS * KNN + en[mt] * KNN + ekk];
        }
    }

    const int rA = BOUND ? (rowA < O ? rowA : O - 1) : rowA;

    v8f acc[MT];
#pragma unroll
    for (int mt = 0; mt < MT; ++mt) acc[mt] = {};

    auto step = [&](int k0) {
        v16bf a;
#pragma unroll
        for (int e = 0; e < 16; ++e) {
            int k = k0 + kbA + e;
            float v;
            if (BOUND) {
                int kc = k < Kd ? k : Kd - 1;
                v = W[(long)rA * Kd + kc];
                v = (rowA < O && k < Kd) ? v : 0.f;
            } else {
                v = W[(long)rowA * Kd + k];
            }
            a[e] = (__bf16)v;
        }
#pragma unroll
        for (int mt = 0; mt < MT; ++mt) {
            const int col = colB + 16 * mt;
            v16bf bb;
#pragma unroll
            for (int e = 0; e < 16; ++e) {
                int k = k0 + kbB + e;
                float v;
                if (BMODE == 0) {
                    if (BOUND) {
                        int kc = k < Kd ? k : Kd - 1;
                        int cc = col < M ? col : M - 1;
                        v = Xb[(long)kc * M + cc];
                        v = (k < Kd && col < M) ? v : 0.f;
                    } else {
                        v = Xb[(long)k * M + col];
                    }
                } else {
                    int kc = BOUND ? (k < Kd ? k : Kd - 1) : k;
                    bool lo = kc < Cin;
                    int  kA = lo ? kc : kc - Cin;        // valid channel either way
                    int  rB = lo ? nbi[mt] : en[mt];
                    float v1 = Xb[(long)kA * NPTS + rB];
                    float v0 = Xb[(long)kA * NPTS + en[mt]];
                    v = lo ? (v1 - v0) : v1;
                    if (BOUND) v = (k < Kd) ? v : 0.f;
                }
                bb[e] = (__bf16)v;
            }
            acc[mt] = __builtin_amdgcn_wmma_f32_16x16x32_bf16(false, a, false, bb,
                                                              (short)0, acc[mt], false, false);
        }
    };

    for (int k0 = 0; k0 < Kd; k0 += 32) {
        if (BMODE == 0 && !BOUND && k0 + 64 <= Kd)
            __builtin_prefetch(&Xb[(long)(k0 + 32 + kbB) * M + colB], 0, 1);
        step(k0);
    }

    float* Ob = Out + (long)b * outBatchStride;
#pragma unroll
    for (int mt = 0; mt < MT; ++mt) {
        const int col = colB + 16 * mt;
#pragma unroll
        for (int rr = 0; rr < 8; ++rr) {
            int orow = (to << 4) + rr + 8 * hi;
            float v = acc[mt][rr];
            if (EPI == 1) {
                int oc = BOUND ? (orow < O ? orow : O - 1) : orow;
                v = lrelu_f(v * scale[oc] + shift[oc]);
            } else if (EPI == 2) {
                int oc = BOUND ? (orow < O ? orow : O - 1) : orow;
                v += scale[oc];
            }
            if (!BOUND || (orow < O && col < M))
                Ob[(long)orow * M + col] = v;
        }
    }
}

// ---------------- TDM contiguous copy: global -> LDS -> global ----------------
// One wave per 16 KB chunk. D# per CDNA5 ISA:
// group0: count=1, lds_addr, global_addr, type=2; group1: data_size=4B,
// tensor_dim0 = remaining elements (clamps the tail via TDM OOB rules),
// tensor_dim1 = 1, tile = 4096x1, stride0 = 4096. Groups 2/3/4 zero (<=2D tensor).
// This toolchain (clang-23) uses the 6-arg builtin form.
__global__ void tdm_copy_k(const float* __restrict__ src, float* __restrict__ dst,
                           long nElemPerBatch, long srcBatchStride, long dstBatchStride,
                           int chunksPerBatch) {
    __shared__ float buf[TDM_CHUNK];
    int cid = blockIdx.x % chunksPerBatch;
    int b   = blockIdx.x / chunksPerBatch;
    long off = (long)cid * TDM_CHUNK;
    unsigned rem = (unsigned)(nElemPerBatch - off);
    const float* s = src + (long)b * srcBatchStride + off;
    float*       d = dst + (long)b * dstBatchStride + off;

    unsigned lds   = (unsigned)(size_t)buf;            // LDS byte offset (addr[31:0])
    unsigned long long gs = (unsigned long long)s;
    unsigned long long gd = (unsigned long long)d;

    v4u g0l = { 1u, lds, (unsigned)gs, (unsigned)((gs >> 32) & 0x01FFFFFFu) | (2u << 30) };
    v4u g0s = { 1u, lds, (unsigned)gd, (unsigned)((gd >> 32) & 0x01FFFFFFu) | (2u << 30) };
    v8i g1;
    g1[0] = 0x20000;                                   // data_size = 2 (4 bytes)
    g1[1] = (int)((rem & 0xFFFFu) << 16);              // tensor_dim0 [63:48]
    g1[2] = (int)(((rem >> 16) & 0xFFFFu) | (1u << 16)); // dim0 hi | tensor_dim1=1 lo
    g1[3] = (int)(TDM_CHUNK << 16);                    // dim1 hi=0 | tile_dim0
    g1[4] = 1;                                         // tile_dim1 = 1, tile_dim2 = 0
    g1[5] = TDM_CHUNK;                                 // tensor_dim0_stride lo
    g1[6] = 0;
    g1[7] = 0;
    v4i z4 = { 0, 0, 0, 0 };
    v8i z8 = { 0, 0, 0, 0, 0, 0, 0, 0 };

    __builtin_amdgcn_tensor_load_to_lds(g0l, g1, z4, z4, z8, 0);
    __builtin_amdgcn_s_wait_tensorcnt(0);
    __builtin_amdgcn_tensor_store_from_lds(g0s, g1, z4, z4, z8, 0);
    __builtin_amdgcn_s_wait_tensorcnt(0);
}

// ---------------- reductions ----------------
__global__ void max_over_k_k(const float* __restrict__ X, float* __restrict__ Y, long total) {
    long i = (long)blockIdx.x * blockDim.x + threadIdx.x;   // over B*C*NPTS
    if (i >= total) return;
    const float* p = X + i * KNN;
    float m = p[0];
#pragma unroll
    for (int k = 1; k < KNN; ++k) m = fmaxf(m, p[k]);
    Y[i] = m;
}

__global__ void maxn_broadcast_k(const float* __restrict__ X, float* __restrict__ dst,
                                 int C, int Cd, int c0) {
    int bc = blockIdx.x, b = bc / C, c = bc % C;
    const float* p = X + ((long)b * C + c) * NPTS;
    float m = -3.4e38f;
    for (int n = threadIdx.x; n < NPTS; n += blockDim.x) m = fmaxf(m, p[n]);
    __shared__ float s[256];
    s[threadIdx.x] = m; __syncthreads();
    for (int st = 128; st > 0; st >>= 1) {
        if (threadIdx.x < st) s[threadIdx.x] = fmaxf(s[threadIdx.x], s[threadIdx.x + st]);
        __syncthreads();
    }
    float mm = s[0];
    float* d = dst + ((long)b * Cd + c0 + c) * NPTS;
    for (int n = threadIdx.x; n < NPTS; n += blockDim.x) d[n] = mm;
}

__global__ void maxmean_n_k(const float* __restrict__ X, float* __restrict__ dst, int C) {
    int bc = blockIdx.x, b = bc / C, c = bc % C;
    const float* p = X + ((long)b * C + c) * NPTS;
    float m = -3.4e38f, su = 0.f;
    for (int n = threadIdx.x; n < NPTS; n += blockDim.x) { float v = p[n]; m = fmaxf(m, v); su += v; }
    __shared__ float sm[256], ss[256];
    sm[threadIdx.x] = m; ss[threadIdx.x] = su; __syncthreads();
    for (int st = 128; st > 0; st >>= 1) {
        if (threadIdx.x < st) {
            sm[threadIdx.x] = fmaxf(sm[threadIdx.x], sm[threadIdx.x + st]);
            ss[threadIdx.x] += ss[threadIdx.x + st];
        }
        __syncthreads();
    }
    if (threadIdx.x == 0) {
        dst[(long)b * 2 * C + c]     = sm[0];
        dst[(long)b * 2 * C + C + c] = ss[0] * (1.f / NPTS);
    }
}

// ---------------- host-side launchers ----------------
template <int BMODE, int EPI, bool BOUND>
static inline void run_gemm(hipStream_t st, const float* W, const float* X, const int* idx,
                            float* out, int O, int Kd, int M, int Cin,
                            long xbs, long obs, const float* sc, const float* sh) {
    int tilesO = (O + 15) / 16, tilesM = (M + 15) / 16;
    if (BOUND) {
        long waves = (long)NB * tilesO * tilesM;
        unsigned blocks = (unsigned)((waves + 7) / 8);
        gemm_bf16_k<BMODE, EPI, true, 1><<<blocks, 256, 0, st>>>(W, X, idx, out, O, Kd, M, Cin,
                                                                 xbs, obs, sc, sh);
    } else {  // all unbounded call sites have tilesM % 4 == 0
        long waves = (long)NB * tilesO * (tilesM / 4);
        unsigned blocks = (unsigned)((waves + 7) / 8);
        gemm_bf16_k<BMODE, EPI, false, 4><<<blocks, 256, 0, st>>>(W, X, idx, out, O, Kd, M, Cin,
                                                                  xbs, obs, sc, sh);
    }
}

static inline void tdm_copy(hipStream_t st, const float* src, float* dst,
                            long nElem, long sStride, long dStride) {
    int chunks = (int)((nElem + TDM_CHUNK - 1) / TDM_CHUNK);
    tdm_copy_k<<<NB * chunks, 32, 0, st>>>(src, dst, nElem, sStride, dStride, chunks);
}

extern "C" void kernel_launch(void* const* d_in, const int* in_sizes, int n_in,
                              void* d_out, int out_size, void* d_ws, size_t ws_size,
                              hipStream_t stream) {
    // -------- input mapping (setup_inputs flattening order) --------
    const float* x     = (const float*)d_in[0];
    const float* w1    = (const float*)d_in[1];
    const float* bn1[4]= {(const float*)d_in[2],(const float*)d_in[3],(const float*)d_in[4],(const float*)d_in[5]};
    const float* w2    = (const float*)d_in[6];
    const float* bn2[4]= {(const float*)d_in[7],(const float*)d_in[8],(const float*)d_in[9],(const float*)d_in[10]};
    const float* w3    = (const float*)d_in[11];
    const float* bn3[4]= {(const float*)d_in[12],(const float*)d_in[13],(const float*)d_in[14],(const float*)d_in[15]};
    const float* w4    = (const float*)d_in[16];
    const float* bn4[4]= {(const float*)d_in[17],(const float*)d_in[18],(const float*)d_in[19],(const float*)d_in[20]};
    const float* w5    = (const float*)d_in[21];
    const float* bn5[4]= {(const float*)d_in[22],(const float*)d_in[23],(const float*)d_in[24],(const float*)d_in[25]};
    const float* w6    = (const float*)d_in[26];
    const float* w7    = (const float*)d_in[27];
    const float* w8    = (const float*)d_in[28];
    const float* bn6[4]= {(const float*)d_in[29],(const float*)d_in[30],(const float*)d_in[31],(const float*)d_in[32]};
    const float* w9    = (const float*)d_in[33];
    const float* bn7[4]= {(const float*)d_in[34],(const float*)d_in[35],(const float*)d_in[36],(const float*)d_in[37]};
    const float* w10   = (const float*)d_in[38];
    const float* lin1w = (const float*)d_in[39];
    const float* lin2w = (const float*)d_in[40];
    const float* lin2b = (const float*)d_in[41];
    const float* lin3w = (const float*)d_in[42];
    const float* lin3b = (const float*)d_in[43];
    float* out = (float*)d_out;

    // -------- workspace bump allocator --------
    char* base = (char*)d_ws;
    size_t cur = 0;
    auto alloc = [&](size_t nfl) -> float* {
        float* p = (float*)(base + cur);
        cur += ((nfl * sizeof(float) + 255) & ~(size_t)255);
        return p;
    };
    float *s1 = alloc(64),  *h1 = alloc(64);
    float *s2 = alloc(64),  *h2 = alloc(64);
    float *s3 = alloc(128), *h3 = alloc(128);
    float *s4 = alloc(256), *h4 = alloc(256);
    float *s5 = alloc(1024),*h5 = alloc(1024);
    float *s6 = alloc(512), *h6 = alloc(512);
    float *s7 = alloc(256), *h7 = alloc(256), *h7f = alloc(256);
    float *gram = alloc((size_t)NB * NPTS * NPTS);
    float *xx   = alloc((size_t)NB * NPTS);
    int   *idxb = (int*)alloc((size_t)NB * NPTS * KNN);
    float *nk256 = alloc((size_t)NB * 256 * NKM);
    float *nk64a = alloc((size_t)NB * 64 * NKM);
    float *nk64b = alloc((size_t)NB * 64 * NKM);
    float *x1m = alloc((size_t)NB * 64 * NPTS);
    float *y1  = alloc((size_t)NB * 64 * NPTS);
    float *x2  = alloc((size_t)NB * 64 * NPTS);
    float *y2  = alloc((size_t)NB * 64 * NPTS);
    float *y3  = alloc((size_t)NB * 64 * NPTS);
    float *x3  = alloc((size_t)NB * 128 * NPTS);
    float *x4  = alloc((size_t)NB * 256 * NPTS);
    float *xs1 = alloc((size_t)NB * 512 * NPTS);
    float *ys1 = alloc((size_t)NB * 192 * NPTS);
    float *yg  = alloc((size_t)NB * 1024 * NPTS);
    float *ys2 = alloc((size_t)NB * 1216 * NPTS);
    float *x5  = alloc((size_t)NB * 1024 * NPTS);
    float *xs2 = alloc((size_t)NB * 2048);
    float *hb1 = alloc((size_t)NB * 512);
    float *hb2 = alloc((size_t)NB * 256);
    float *sb1 = alloc((size_t)NB * 512 * NPTS);
    float *sb2 = alloc((size_t)NB * 256 * NPTS);

    auto g1 = [](long n) { return dim3((unsigned)((n + 255) / 256)); };

    // -------- BN scale/shift folding --------
    bn_prep_k<<<1, 64,   0, stream>>>(bn1[0], bn1[1], bn1[2], bn1[3], s1, h1, 64);
    bn_prep_k<<<1, 64,   0, stream>>>(bn2[0], bn2[1], bn2[2], bn2[3], s2, h2, 64);
    bn_prep_k<<<1, 128,  0, stream>>>(bn3[0], bn3[1], bn3[2], bn3[3], s3, h3, 128);
    bn_prep_k<<<1, 256,  0, stream>>>(bn4[0], bn4[1], bn4[2], bn4[3], s4, h4, 256);
    bn_prep_k<<<4, 256,  0, stream>>>(bn5[0], bn5[1], bn5[2], bn5[3], s5, h5, 1024);
    bn_prep_k<<<2, 256,  0, stream>>>(bn6[0], bn6[1], bn6[2], bn6[3], s6, h6, 512);
    bn_prep_k<<<1, 256,  0, stream>>>(bn7[0], bn7[1], bn7[2], bn7[3], s7, h7, 256);
    fold_bias_k<<<1, 256, 0, stream>>>(s7, h7, lin2b, h7f, 256);

    auto knn = [&](const float* src, int C) {
        long waves = (long)NB * 128 * 128;
        gram_wmma_k<<<(unsigned)((waves + 7) / 8), 256, 0, stream>>>(src, gram, C);
        gram_diag_k<<<g1((long)NB * NPTS), 256, 0, stream>>>(gram, xx);
        knn_topk_k<<<NB * NPTS, 32, 0, stream>>>(gram, xx, idxb);
    };

    // ---- stage 1: x1 = block(edge(x), w1, bn1); y = block(x1, w6, bn2)
    knn(x, 3);
    run_gemm<1, 1, true >(stream, w1, x, idxb, nk64a, 64, 6, NKM, 3, 3L * NPTS, 64L * NKM, s1, h1);
    run_gemm<0, 1, false>(stream, w6, nk64a, nullptr, nk64b, 64, 64, NKM, 0, 64L * NKM, 64L * NKM, s2, h2);
    max_over_k_k<<<g1((long)NB * 64 * NPTS), 256, 0, stream>>>(nk64a, x1m, (long)NB * 64 * NPTS);
    max_over_k_k<<<g1((long)NB * 64 * NPTS), 256, 0, stream>>>(nk64b, y1,  (long)NB * 64 * NPTS);

    // ---- x2 = max_k block(edge(x1m), w2, bn2)
    knn(x1m, 64);
    run_gemm<1, 1, false>(stream, w2, x1m, idxb, nk64b, 64, 128, NKM, 64, 64L * NPTS, 64L * NKM, s2, h2);
    max_over_k_k<<<g1((long)NB * 64 * NPTS), 256, 0, stream>>>(nk64b, x2, (long)NB * 64 * NPTS);

    // ---- y2 = max_k block(block(edge(y1), w2, bn2), w6, bn2)
    knn(y1, 64);
    run_gemm<1, 1, false>(stream, w2, y1, idxb, nk64a, 64, 128, NKM, 64, 64L * NPTS, 64L * NKM, s2, h2);
    run_gemm<0, 1, false>(stream, w6, nk64a, nullptr, nk64b, 64, 64, NKM, 0, 64L * NKM, 64L * NKM, s2, h2);
    max_over_k_k<<<g1((long)NB * 64 * NPTS), 256, 0, stream>>>(nk64b, y2, (long)NB * 64 * NPTS);

    // ---- x3 = max_k block(edge(x2), w3, bn3)
    knn(x2, 64);
    run_gemm<1, 1, false>(stream, w3, x2, idxb, nk256, 128, 128, NKM, 64, 64L * NPTS, 128L * NKM, s3, h3);
    max_over_k_k<<<g1((long)NB * 128 * NPTS), 256, 0, stream>>>(nk256, x3, (long)NB * 128 * NPTS);

    // ---- y3 = max_k block(edge(y2), w2, bn2)
    knn(y2, 64);
    run_gemm<1, 1, false>(stream, w2, y2, idxb, nk64a, 64, 128, NKM, 64, 64L * NPTS, 64L * NKM, s2, h2);
    max_over_k_k<<<g1((long)NB * 64 * NPTS), 256, 0, stream>>>(nk64a, y3, (long)NB * 64 * NPTS);

    // ---- x4 = max_k block(edge(x3), w4, bn4)
    knn(x3, 128);
    run_gemm<1, 1, false>(stream, w4, x3, idxb, nk256, 256, 256, NKM, 128, 128L * NPTS, 256L * NKM, s4, h4);
    max_over_k_k<<<g1((long)NB * 256 * NPTS), 256, 0, stream>>>(nk256, x4, (long)NB * 256 * NPTS);

    // ---- concat via TDM DMA copies: xs1 = [x1m, x2, x3, x4]; ys1 = [y1, y2, y3]
    tdm_copy(stream, x1m, xs1 + 0L   * NPTS,  64L * NPTS,  64L * NPTS, 512L * NPTS);
    tdm_copy(stream, x2,  xs1 + 64L  * NPTS,  64L * NPTS,  64L * NPTS, 512L * NPTS);
    tdm_copy(stream, x3,  xs1 + 128L * NPTS, 128L * NPTS, 128L * NPTS, 512L * NPTS);
    tdm_copy(stream, x4,  xs1 + 256L * NPTS, 256L * NPTS, 256L * NPTS, 512L * NPTS);
    tdm_copy(stream, y1,  ys1 + 0L   * NPTS,  64L * NPTS,  64L * NPTS, 192L * NPTS);
    tdm_copy(stream, y2,  ys1 + 64L  * NPTS,  64L * NPTS,  64L * NPTS, 192L * NPTS);
    tdm_copy(stream, y3,  ys1 + 128L * NPTS,  64L * NPTS,  64L * NPTS, 192L * NPTS);

    // ---- yg = block(ys1, w7, bn5); ys2 = [bcast(max_n yg), y1, y2, y3]
    run_gemm<0, 1, false>(stream, w7, ys1, nullptr, yg, 1024, 192, NPTS, 0, 192L * NPTS, 1024L * NPTS, s5, h5);
    maxn_broadcast_k<<<NB * 1024, 256, 0, stream>>>(yg, ys2, 1024, 1216, 0);
    tdm_copy(stream, y1, ys2 + 1024L * NPTS, 64L * NPTS, 64L * NPTS, 1216L * NPTS);
    tdm_copy(stream, y2, ys2 + 1088L * NPTS, 64L * NPTS, 64L * NPTS, 1216L * NPTS);
    tdm_copy(stream, y3, ys2 + 1152L * NPTS, 64L * NPTS, 64L * NPTS, 1216L * NPTS);

    // ---- x5 = block(xs1, w5, bn5); xs2 = [max_n, mean_n]
    run_gemm<0, 1, false>(stream, w5, xs1, nullptr, x5, 1024, 512, NPTS, 0, 512L * NPTS, 1024L * NPTS, s5, h5);
    maxmean_n_k<<<NB * 1024, 256, 0, stream>>>(x5, xs2, 1024);

    // ---- classification head (tiny M=1 GEMMs -> bounded path)
    run_gemm<0, 1, true>(stream, lin1w, xs2, nullptr, hb1, 512, 2048, 1, 0, 2048L, 512L, s6, h6);
    run_gemm<0, 1, true>(stream, lin2w, hb1, nullptr, hb2, 256, 512, 1, 0, 512L, 256L, s7, h7f);
    run_gemm<0, 2, true>(stream, lin3w, hb2, nullptr, out, 5, 256, 1, 0, 256L, 5L, lin3b, nullptr);

    // ---- segmentation head
    run_gemm<0, 1, false>(stream, w8, ys2, nullptr, sb1, 512, 1216, NPTS, 0, 1216L * NPTS, 512L * NPTS, s6, h6);
    run_gemm<0, 1, false>(stream, w9, sb1, nullptr, sb2, 256, 512, NPTS, 0, 512L * NPTS, 256L * NPTS, s7, h7);
    run_gemm<0, 0, true >(stream, w10, sb2, nullptr, out + 20, 7, 256, NPTS, 0, 256L * NPTS, 7L * NPTS, nullptr, nullptr);
}
